// MultiHeadCrossAttention_11854109737409
// MI455X (gfx1250) — matrix operand order repaired and checked by reading, not compiled, not run on hardware
//
#include <hip/hip_runtime.h>

typedef _Float16 h16;
typedef __attribute__((ext_vector_type(4)))  _Float16 v4h;
typedef __attribute__((ext_vector_type(8)))  _Float16 v8h;
typedef __attribute__((ext_vector_type(16))) _Float16 v16h;
typedef __attribute__((ext_vector_type(8)))  float    v8f;

#define BB  2
#define SS  2048
#define DD  1024
#define HH  16
#define HDD 64

// ---------------------------------------------------------------- WMMA core

__device__ __forceinline__ v8f wmma_f16(v16h a, v16h b, v8f c) {
  // D = A(16x32 f16) * B(32x16 f16) + C(16x16 f32)
  return __builtin_amdgcn_wmma_f32_16x16x32_f16(false, a, false, b,
                                                (short)0, c, false, false);
}

__device__ __forceinline__ v8f zero8() {
  v8f z;
#pragma unroll
  for (int i = 0; i < 8; ++i) z[i] = 0.0f;
  return z;
}

__device__ __forceinline__ void lds_wave_fence() {
#if __has_builtin(__builtin_amdgcn_s_wait_dscnt)
  __builtin_amdgcn_s_wait_dscnt(0);
#endif
  __asm__ volatile("" ::: "memory");
}

// A fragment, 16x32 f16, row-major source with leading dim lda (elements).
// ISA 7.12.2: lanes 0-15 row M=lane, halves 0-7 -> K 0-7, 8-15 -> K 16-23;
// lanes 16-31 row M=lane-16, halves 0-7 -> K 8-15, 8-15 -> K 24-31.
__device__ __forceinline__ v16h load_A_frag(const h16* p0, int lda) {
  const int lane = threadIdx.x & 31;
  const int row  = lane & 15;
  const int kb   = (lane >> 4) << 3;          // 0 or 8
  const h16* p   = p0 + (size_t)row * lda;
  v8h lo = *(const v8h*)(p + kb);             // K = kb .. kb+7
  v8h hi = *(const v8h*)(p + kb + 16);        // K = kb+16 .. kb+23
  v16h r;
#pragma unroll
  for (int i = 0; i < 8; ++i) { r[i] = lo[i]; r[i + 8] = hi[i]; }
  return r;
}

// B fragment, 32x16: lane holds column n = lane&15, K rows kb..kb+7 and
// kb+16..kb+23 with kb = 8*(lane>>4).  Source stores B transposed: element (k,n) at
// p0[n*ld + k], K contiguous -> two b128 loads per lane.
__device__ __forceinline__ v16h load_B_frag_kt(const h16* p0, int ld) {
  const int lane = threadIdx.x & 31;
  const int n    = lane & 15;
  const int k0   = (lane >> 4) << 3;          // 0 or 8
  const h16* p   = p0 + (size_t)n * ld + k0;
  v8h lo = *(const v8h*)(p);
  v8h hi = *(const v8h*)(p + 16);
  v16h r;
#pragma unroll
  for (int i = 0; i < 8; ++i) { r[i] = lo[i]; r[i + 8] = hi[i]; }
  return r;
}

// ---------------------------------------------------------------- conversions

__global__ __launch_bounds__(256) void cvt_f32_f16_kernel(
    const float* __restrict__ s, h16* __restrict__ d, int n4) {
  int i = blockIdx.x * blockDim.x + threadIdx.x;
  if (i >= n4) return;
  const float4 v = ((const float4*)s)[i];
  v4h o;
  o[0] = (h16)v.x; o[1] = (h16)v.y; o[2] = (h16)v.z; o[3] = (h16)v.w;
  *(v4h*)(d + (size_t)i * 4) = o;
}

// src [K,N] f32 -> dst [N,K] f16, 32x32 tiles through LDS (coalesced both sides)
__global__ __launch_bounds__(256) void cvt_transpose_kernel(
    const float* __restrict__ src, h16* __restrict__ dst, int K, int N) {
  __shared__ float tile[32][33];
  const int n0 = blockIdx.x * 32;
  const int k0 = blockIdx.y * 32;
  const int tx = threadIdx.x & 31;
  const int ty = threadIdx.x >> 5;            // 0..7
#pragma unroll
  for (int r = ty; r < 32; r += 8)
    tile[r][tx] = src[(size_t)(k0 + r) * N + n0 + tx];
  __syncthreads();
#pragma unroll
  for (int r = ty; r < 32; r += 8)
    dst[(size_t)(n0 + r) * K + k0 + tx] = (h16)tile[tx][r];
}

// ---------------------------------------------------------------- WMMA GEMM
// C[M,N] = A[M,K](f16 row-major) * Bt[N,K](f16, B transposed) + bias[N]
// Block: 256 thr (8 waves); each wave computes 32(M) x 64(N); block = 256 x 64.
// Per k-step (32): 2 A-frags + 4 B-frags (12 x b128 loads), 8 WMMAs.
// mode 0: f32 out -> outF[M,N]
// mode 1: kv scatter: n -> h=n>>7, j=n&127; j<64 -> K[b,h,s,j] else Vt[b,h,j-64,s]
// mode 2: q scatter:  n -> h=n>>6, j=n&63  -> Q[b,h,s,j]

__global__ __launch_bounds__(256) void gemm_wmma_kernel(
    const h16* __restrict__ A, const h16* __restrict__ Bt,
    const float* __restrict__ bias,
    float* __restrict__ outF, h16* __restrict__ out0, h16* __restrict__ out1,
    int M, int N, int K, int S, int mode) {
  const int wave  = threadIdx.x >> 5;
  const int lane  = threadIdx.x & 31;
  const int half  = lane >> 4;
  const int nl    = lane & 15;
  const int nbase = blockIdx.x * 64;
  const int row0  = blockIdx.y * 256 + wave * 32;

  v8f acc[2][4];
#pragma unroll
  for (int g = 0; g < 2; ++g)
#pragma unroll
    for (int f = 0; f < 4; ++f) acc[g][f] = zero8();

  const h16* Arow0 = A + (size_t)row0 * K;
  const h16* Arow1 = A + (size_t)(row0 + 16) * K;
  const h16* Bcol  = Bt + (size_t)nbase * K;

  for (int kk = 0; kk < K; kk += 32) {
    if (kk + 32 < K) {
      __builtin_prefetch(Arow0 + kk + 32, 0, 1);
      __builtin_prefetch(Arow1 + kk + 32, 0, 1);
    }
    const v16h a0 = load_A_frag(Arow0 + kk, K);
    const v16h a1 = load_A_frag(Arow1 + kk, K);
#pragma unroll
    for (int f = 0; f < 4; ++f) {
      const v16h bf = load_B_frag_kt(Bcol + (size_t)(f * 16) * K + kk, K);
      acc[0][f] = wmma_f16(a0, bf, acc[0][f]);
      acc[1][f] = wmma_f16(a1, bf, acc[1][f]);
    }
  }

#pragma unroll
  for (int g = 0; g < 2; ++g) {
#pragma unroll
    for (int f = 0; f < 4; ++f) {
#pragma unroll
      for (int i = 0; i < 8; ++i) {
        const int m = row0 + g * 16 + i + 8 * half;  // C: M = vgpr + 8*half
        const int n = nbase + f * 16 + nl;           //    N = lane&15
        const float v = acc[g][f][i] + bias[n];
        if (mode == 0) {
          outF[(size_t)m * N + n] = v;
        } else if (mode == 1) {
          const int bb = m / S, ss = m % S;
          const int hh = n >> 7, jj = n & 127;
          const h16 hv = (h16)v;
          if (jj < HDD)   // K: [B,H,S,64]
            out0[(((size_t)bb * HH + hh) * S + ss) * HDD + jj] = hv;
          else            // V transposed: [B,H,64,S]
            out1[(((size_t)bb * HH + hh) * HDD + (jj - HDD)) * S + ss] = hv;
        } else {
          const int bb = m / S, ss = m % S;
          const int hh = n >> 6, jj = n & 63;
          out0[(((size_t)bb * HH + hh) * S + ss) * HDD + jj] = (h16)v;
        }
      }
    }
  }
}

// ------------------------------------------------------------ Flash attention
// Q,K: [B,H,S,64] f16.  Vt: [B,H,64,S] f16.  Out: values [B,S,D] f16.
// One wave per 16-row Q tile, 8 waves/block. Per 32-key chunk:
// 4 score WMMAs + online softmax (shuffle reductions within 16-lane halves)
// + P f32->f16 through wave-private LDS + 4 P@V WMMAs (all b128 fragment loads).

__global__ __launch_bounds__(256) void attn_wmma_kernel(
    const h16* __restrict__ Q, const h16* __restrict__ Kt,
    const h16* __restrict__ Vt, h16* __restrict__ O, int S) {
  __shared__ __align__(16) h16 Pls[8][16 * 32];

  const int wave = threadIdx.x >> 5;
  const int lane = threadIdx.x & 31;
  const int half = lane >> 4;
  const int nl   = lane & 15;
  const int h    = blockIdx.y;
  const int b    = blockIdx.z;
  const size_t bh = (size_t)b * HH + h;

  const h16* Qp  = Q  + bh * S * HDD;
  const h16* Kp  = Kt + bh * S * HDD;
  const h16* VTp = Vt + bh * HDD * S;        // [64, S]
  const int  q0  = blockIdx.x * 128 + wave * 16;

  const v16h qa0 = load_A_frag(Qp + (size_t)q0 * HDD, HDD);        // hd 0..31
  const v16h qa1 = load_A_frag(Qp + (size_t)q0 * HDD + 32, HDD);   // hd 32..63

  v8f o[4];
#pragma unroll
  for (int f = 0; f < 4; ++f) o[f] = zero8();

  float mrow[8], lrow[8];
#pragma unroll
  for (int i = 0; i < 8; ++i) { mrow[i] = -1e30f; lrow[i] = 0.0f; }

  const float c = 0.125f * 1.44269504088896f;  // 1/sqrt(64) * log2(e)
  h16* P = Pls[wave];

  for (int kc = 0; kc < S; kc += 32) {
    // ---- scores: two 16x16 tiles covering keys kc..kc+31
    v8f s0 = zero8(), s1 = zero8();
    {
      // K stored [key, hd]: for score B-frag (k=hd, n=key) it's already
      // transposed-with-k-contiguous -> b128 loads.
      v16h b00 = load_B_frag_kt(Kp + (size_t)kc * HDD, HDD);
      v16h b01 = load_B_frag_kt(Kp + (size_t)kc * HDD + 32, HDD);
      s0 = wmma_f16(qa0, b00, s0);
      s0 = wmma_f16(qa1, b01, s0);
      v16h b10 = load_B_frag_kt(Kp + (size_t)(kc + 16) * HDD, HDD);
      v16h b11 = load_B_frag_kt(Kp + (size_t)(kc + 16) * HDD + 32, HDD);
      s1 = wmma_f16(qa0, b10, s1);
      s1 = wmma_f16(qa1, b11, s1);
    }

    // ---- online softmax (row = vgpr + 8*half spans the 16 lanes of a half)
    float resc[8];
#pragma unroll
    for (int i = 0; i < 8; ++i) {
      const float t0 = s0[i] * c;
      const float t1 = s1[i] * c;
      float a = fmaxf(t0, t1);
#pragma unroll
      for (int d = 1; d < 16; d <<= 1) a = fmaxf(a, __shfl_xor(a, d, 16));
      const float mnew = fmaxf(mrow[i], a);
      const float r    = exp2f(mrow[i] - mnew);
      const float p0   = exp2f(t0 - mnew);
      const float p1   = exp2f(t1 - mnew);
      float ps = p0 + p1;
#pragma unroll
      for (int d = 1; d < 16; d <<= 1) ps += __shfl_xor(ps, d, 16);
      lrow[i] = lrow[i] * r + ps;
      mrow[i] = mnew;
      resc[i] = r;
      const int rowi = i + 8 * half;
      P[rowi * 32 + nl]      = (h16)p0;      // key column kc + nl
      P[rowi * 32 + nl + 16] = (h16)p1;      // key column kc + 16 + nl
    }

    // ---- rescale running output
#pragma unroll
    for (int f = 0; f < 4; ++f)
#pragma unroll
      for (int i = 0; i < 8; ++i) o[f][i] *= resc[i];

    // ---- P@V (DS ops are in-order within a wave; fence for the compiler)
    lds_wave_fence();
    v16h pa = load_A_frag(P, 32);
#pragma unroll
    for (int f = 0; f < 4; ++f) {
      // Vt [64, S]: B-frag (k=key, n=hd col f*16+nl), key contiguous -> b128
      v16h vb = load_B_frag_kt(VTp + (size_t)(f * 16) * S + kc, S);
      o[f] = wmma_f16(pa, vb, o[f]);
    }
    lds_wave_fence();
  }

  // ---- epilogue: O / l, store as f16 values[b, s, h*64 + col]
#pragma unroll
  for (int f = 0; f < 4; ++f)
#pragma unroll
    for (int i = 0; i < 8; ++i) {
      const int m   = q0 + i + 8 * half;
      const int col = h * HDD + f * 16 + nl;
      O[((size_t)b * S + m) * DD + col] = (h16)(o[f][i] / lrow[i]);
    }
}

// ---------------------------------------------------------------- dispatcher

extern "C" void kernel_launch(void* const* d_in, const int* in_sizes, int n_in,
                              void* d_out, int out_size, void* d_ws, size_t ws_size,
                              hipStream_t stream) {
  (void)in_sizes; (void)n_in; (void)out_size; (void)ws_size;

  const float* x   = (const float*)d_in[0];
  const float* y   = (const float*)d_in[1];
  const float* Wkv = (const float*)d_in[2];
  const float* bkv = (const float*)d_in[3];
  const float* Wq  = (const float*)d_in[4];
  const float* bq  = (const float*)d_in[5];
  const float* Wo  = (const float*)d_in[6];
  const float* bo  = (const float*)d_in[7];
  float* out = (float*)d_out;

  const size_t nX   = (size_t)BB * SS * DD;        // 4,194,304
  const size_t nWkv = (size_t)DD * 2 * DD;
  const size_t nWq  = (size_t)DD * DD;

  h16* w = (h16*)d_ws;
  h16* x16   = w;  w += nX;
  h16* y16   = w;  w += nX;
  h16* wkvT  = w;  w += nWkv;                      // [2D, D]  (W_kv^T)
  h16* wqT   = w;  w += nWq;                       // [D, D]   (W_q^T)
  h16* woT   = w;  w += nWq;                       // [D, D]   (W_o^T)
  h16* k16   = w;  w += nX;                        // [B,H,S,64]
  h16* vT16  = w;  w += nX;                        // [B,H,64,S]
  h16* q16   = w;  w += nX;                        // [B,H,S,64]
  h16* val16 = x16;                                // reuse: x16 dead after kv GEMM

  // f32 -> f16 conversions (weights transposed for K-contiguous B-fragments)
  cvt_f32_f16_kernel<<<nX / 4 / 256, 256, 0, stream>>>(x, x16, (int)(nX / 4));
  cvt_f32_f16_kernel<<<nX / 4 / 256, 256, 0, stream>>>(y, y16, (int)(nX / 4));
  cvt_transpose_kernel<<<dim3(2 * DD / 32, DD / 32), 256, 0, stream>>>(Wkv, wkvT, DD, 2 * DD);
  cvt_transpose_kernel<<<dim3(DD / 32, DD / 32), 256, 0, stream>>>(Wq, wqT, DD, DD);
  cvt_transpose_kernel<<<dim3(DD / 32, DD / 32), 256, 0, stream>>>(Wo, woT, DD, DD);

  const int M = BB * SS;  // 4096

  // kv = x @ W_kv + b_kv -> K [B,H,S,64], V^T [B,H,64,S]
  gemm_wmma_kernel<<<dim3((2 * DD) / 64, M / 256), 256, 0, stream>>>(
      x16, wkvT, bkv, nullptr, k16, vT16, M, 2 * DD, DD, SS, 1);

  // q = y @ W_q + b_q -> Q [B,H,S,64]
  gemm_wmma_kernel<<<dim3(DD / 64, M / 256), 256, 0, stream>>>(
      y16, wqT, bq, nullptr, q16, nullptr, M, DD, DD, SS, 2);

  // flash attention -> values [B,S,D] f16
  attn_wmma_kernel<<<dim3(SS / 128, HH, BB), 256, 0, stream>>>(
      q16, k16, vT16, val16, SS);

  // out = values @ W_o + b_o (f32)
  gemm_wmma_kernel<<<dim3(DD / 64, M / 256), 256, 0, stream>>>(
      val16, woT, bo, out, nullptr, nullptr, M, DD, DD, SS, 0);
}
